// MultiHeadAttention_75307956568752
// MI455X (gfx1250) — matrix operand order repaired and checked by reading, not compiled, not run on hardware
//
#include <hip/hip_runtime.h>
#include <hip/hip_bf16.h>

typedef unsigned short u16;
typedef unsigned int   u32;
typedef __attribute__((ext_vector_type(16))) __bf16 bf16x16;
typedef __attribute__((ext_vector_type(8)))  float  f32x8;
typedef __attribute__((ext_vector_type(4)))  int    v4i;    // native vector (not HIP_vector_type)

#define WMMA_BF16(A, B, C) \
  __builtin_amdgcn_wmma_f32_16x16x32_bf16(false, (A), false, (B), (short)0, (C), false, false)

#if __has_builtin(__builtin_amdgcn_global_load_async_to_lds_b128) && \
    __has_builtin(__builtin_amdgcn_s_wait_asynccnt)
#define HAS_ASYNC_LDS 1
// Builtin signature (from hipcc diagnostics): (AS1 v4i*, AS3 v4i*, imm int, imm int)
// where v4i is a native clang vector of 4 ints. The immediate offset is applied
// to BOTH the global and LDS address per the ISA.
#define ASYNC_CP16(g, l, off)                                             \
  __builtin_amdgcn_global_load_async_to_lds_b128(                         \
      (__attribute__((address_space(1))) v4i*)(void*)(g),                 \
      (__attribute__((address_space(3))) v4i*)(l), (off), 0)
#define ASYNC_WAIT() __builtin_amdgcn_s_wait_asynccnt(0)
#else
#define HAS_ASYNC_LDS 0
#endif

__device__ __forceinline__ f32x8 zero8() {
  f32x8 z;
#pragma unroll
  for (int i = 0; i < 8; ++i) z[i] = 0.0f;
  return z;
}

union Frag  { int4 i2[2]; bf16x16 v; };      // 32B = one wave32 bf16 A/B fragment per lane
union Ld16h { int4 i[2];  __bf16 h[16]; };
union Ld8h  { int4 i;     __bf16 h[8];  };
union Ld8f  { int4 i[2];  float  f[8];  };

// ---------------------------------------------------------------------------
// One-time fp32 -> bf16 conversion (hardware cvt), 8 elements per thread.
// ---------------------------------------------------------------------------
__global__ __launch_bounds__(256) void convert_bf16(
    const float* __restrict__ src, __bf16* __restrict__ dst) {
  const size_t base = ((size_t)blockIdx.x * 256 + threadIdx.x) * 8;
  Ld8f in; Ld8h out;
  in.i[0] = ((const int4*)(src + base))[0];
  in.i[1] = ((const int4*)(src + base))[1];
#pragma unroll
  for (int j = 0; j < 8; ++j) out.h[j] = (__bf16)in.f[j];
  *(int4*)(dst + base) = out.i;
}

// ---------------------------------------------------------------------------
// Shared GEMM tile machinery: 128x128 tile, BK=32, 8 waves x (32x64) each.
// A tile: bf16 row-major copy -> async-to-LDS when available.
// B tile: bf16, transposed into LDS [n][k] for contiguous B fragments.
// ---------------------------------------------------------------------------
__device__ __forceinline__ void load_a_tile(
    const __bf16* __restrict__ A, __bf16 (&At)[128][32],
    int bm, int kt, int tid) {
  const int arow = tid >> 1, ah = (tid & 1) * 16;
  const __bf16* ap = A + (size_t)(bm + arow) * 1024 + kt + ah;
  __bf16* lp = &At[arow][ah];
#if HAS_ASYNC_LDS
  ASYNC_CP16(ap, lp, 0);
  ASYNC_CP16(ap, lp, 16);
#else
  ((int4*)lp)[0] = ((const int4*)ap)[0];
  ((int4*)lp)[1] = ((const int4*)ap)[1];
#endif
}

__device__ __forceinline__ void load_b_tile(
    const __bf16* __restrict__ B, __bf16 (&Bt)[128][32],
    int bn, int kt, int tid) {
  const int bkr = tid >> 3, bck = (tid & 7) * 16;
  const __bf16* bp = B + (size_t)(kt + bkr) * 1024 + bn + bck;
  Ld16h lh;
  lh.i[0] = ((const int4*)bp)[0];
  lh.i[1] = ((const int4*)bp)[1];
#pragma unroll
  for (int j = 0; j < 16; ++j) Bt[bck + j][bkr] = lh.h[j];
  if (kt + 32 < 1024) __builtin_prefetch(bp + 32 * 1024, 0, 3);
}

// ---------------------------------------------------------------------------
// Fused QKV projection: C[z] = Xbf @ Wbf[z], bf16 in, bf16 out.
// ---------------------------------------------------------------------------
__global__ __launch_bounds__(256) void gemm_qkv(
    const __bf16* __restrict__ X,
    const __bf16* __restrict__ Wq, const __bf16* __restrict__ Wk,
    const __bf16* __restrict__ Wv,
    __bf16* __restrict__ Q, __bf16* __restrict__ Ko, __bf16* __restrict__ V) {
  __shared__ __align__(16) __bf16 At[128][32];
  __shared__ __align__(16) __bf16 Bt[128][32];

  const int tid  = threadIdx.x;
  const int lane = tid & 31, w = tid >> 5;
  const int half = lane >> 4, ln = lane & 15;
  const int wm = (w & 3) * 32, wn = (w >> 2) * 64;
  const int bm = blockIdx.y * 128, bn = blockIdx.x * 128;

  const __bf16* B = (blockIdx.z == 0) ? Wq : (blockIdx.z == 1) ? Wk : Wv;
  __bf16*       C = (blockIdx.z == 0) ? Q  : (blockIdx.z == 1) ? Ko : V;

  f32x8 acc[2][4];
#pragma unroll
  for (int i = 0; i < 2; ++i)
#pragma unroll
    for (int j = 0; j < 4; ++j) acc[i][j] = zero8();

  for (int kt = 0; kt < 1024; kt += 32) {
    load_a_tile(X, At, bm, kt, tid);
    load_b_tile(B, Bt, bn, kt, tid);
#if HAS_ASYNC_LDS
    ASYNC_WAIT();
#endif
    __syncthreads();

    Frag fa[2], fb[4];
#pragma unroll
    for (int mt = 0; mt < 2; ++mt) {        // ISA A layout: lane=row, K split by half
      const int m = wm + mt * 16 + ln;
      fa[mt].i2[0] = *(const int4*)&At[m][half * 8];
      fa[mt].i2[1] = *(const int4*)&At[m][16 + half * 8];
    }
#pragma unroll
    for (int nt = 0; nt < 4; ++nt) {        // ISA B layout: lane=col, K split by half
      const int n = wn + nt * 16 + ln;
      fb[nt].i2[0] = *(const int4*)&Bt[n][half * 8];
      fb[nt].i2[1] = *(const int4*)&Bt[n][16 + half * 8];
    }
#pragma unroll
    for (int mt = 0; mt < 2; ++mt)
#pragma unroll
      for (int nt = 0; nt < 4; ++nt)
        acc[mt][nt] = WMMA_BF16(fa[mt].v, fb[nt].v, acc[mt][nt]);
    __syncthreads();
  }

#pragma unroll
  for (int mt = 0; mt < 2; ++mt)
#pragma unroll
    for (int nt = 0; nt < 4; ++nt)
#pragma unroll
      for (int r = 0; r < 8; ++r) {         // C layout: M = r + 8*half, N = ln
        const int grow = bm + wm + mt * 16 + r + 8 * half;
        const int gcol = bn + wn + nt * 16 + ln;
        C[(size_t)grow * 1024 + gcol] = (__bf16)acc[mt][nt][r];
      }
}

// ---------------------------------------------------------------------------
// Cross-head attention: one wave per token. energy = q @ k^T (2x WMMA K=32),
// softmax over 16 heads via shfl within 16-lane halves, out = attn @ v
// (4x WMMA, K padded 16->32 with zeros) via LDS-staged attn + v^T.
// ---------------------------------------------------------------------------
__global__ __launch_bounds__(256) void attn_kernel(
    const __bf16* __restrict__ Q, const __bf16* __restrict__ K,
    const __bf16* __restrict__ V, __bf16* __restrict__ AO) {
  __shared__ __align__(16) __bf16 attnA[8][16][32];  // per wave: [hq][hk(pad32)]
  __shared__ __align__(16) __bf16 vT[8][64][32];     // per wave: [d][hk(pad32)]

  const int tid  = threadIdx.x;
  const int lane = tid & 31, w = tid >> 5;
  const int half = lane >> 4, ln = lane & 15;
  const size_t token = (size_t)blockIdx.x * 8 + w;

  const __bf16* q = Q + token * 1024;   // [16 heads][64 dims], row-major bf16
  const __bf16* k = K + token * 1024;
  const __bf16* v = V + token * 1024;

  // energy = q @ k^T : M=hq(16), N=hk(16), K=d(64) -> two K=32 WMMAs
  f32x8 e = zero8();
#pragma unroll
  for (int c = 0; c < 2; ++c) {
    Frag fq, fk;
    fq.i2[0] = *(const int4*)(q + ln * 64 + c * 32 + half * 8);
    fq.i2[1] = *(const int4*)(q + ln * 64 + c * 32 + 16 + half * 8);
    fk.i2[0] = *(const int4*)(k + ln * 64 + c * 32 + half * 8);
    fk.i2[1] = *(const int4*)(k + ln * 64 + c * 32 + 16 + half * 8);
    e = WMMA_BF16(fq.v, fk.v, e);
  }

  // softmax over hk: row M = r + 8*half lives in one 16-lane group (N = ln)
  const float inv_scale = 0.03125f;  // 1/sqrt(1024)
#pragma unroll
  for (int r = 0; r < 8; ++r) {
    float x = e[r] * inv_scale;
    float m = x;
#pragma unroll
    for (int d = 8; d >= 1; d >>= 1) m = fmaxf(m, __shfl_xor(m, d, 32));
    const float p = __expf(x - m);
    float s = p;
#pragma unroll
    for (int d = 8; d >= 1; d >>= 1) s += __shfl_xor(s, d, 32);
    attnA[w][r + 8 * half][ln]      = (__bf16)(p / s);
    attnA[w][r + 8 * half][16 + ln] = (__bf16)0.0f;   // zero-pad K 16..31
  }

  // v^T into LDS: vT[d][hk], K rows 16..31 zeroed
  {
    const int hk = lane >> 1, dh = (lane & 1) * 32;
#pragma unroll
    for (int part = 0; part < 4; ++part) {
      Ld8h t;
      t.i = *(const int4*)(v + hk * 64 + dh + part * 8);
#pragma unroll
      for (int j = 0; j < 8; ++j) vT[w][dh + part * 8 + j][hk] = t.h[j];
    }
#pragma unroll
    for (int d2 = 0; d2 < 2; ++d2) {
      const int d = lane * 2 + d2;
#pragma unroll
      for (int j = 0; j < 16; ++j) vT[w][d][16 + j] = (__bf16)0.0f;
    }
  }
  // Per-wave LDS regions only; DS ops from the same wave are in-order.

  Frag fa;
  fa.i2[0] = *(const int4*)&attnA[w][ln][half * 8];
  fa.i2[1] = *(const int4*)&attnA[w][ln][16 + half * 8];
#pragma unroll
  for (int t = 0; t < 4; ++t) {
    Frag fv;
    fv.i2[0] = *(const int4*)&vT[w][t * 16 + ln][half * 8];
    fv.i2[1] = *(const int4*)&vT[w][t * 16 + ln][16 + half * 8];
    f32x8 o = zero8();
    o = WMMA_BF16(fa.v, fv.v, o);
#pragma unroll
    for (int r = 0; r < 8; ++r)
      AO[token * 1024 + (size_t)(r + 8 * half) * 64 + t * 16 + ln] = (__bf16)o[r];
  }
}

// ---------------------------------------------------------------------------
// Output projection: Out = AO(bf16) @ Wo(bf16) + bo, fp32 output.
// ---------------------------------------------------------------------------
__global__ __launch_bounds__(256) void gemm_out(
    const __bf16* __restrict__ A, const __bf16* __restrict__ B,
    const float* __restrict__ bias, float* __restrict__ Out) {
  __shared__ __align__(16) __bf16 At[128][32];
  __shared__ __align__(16) __bf16 Bt[128][32];

  const int tid  = threadIdx.x;
  const int lane = tid & 31, w = tid >> 5;
  const int half = lane >> 4, ln = lane & 15;
  const int wm = (w & 3) * 32, wn = (w >> 2) * 64;
  const int bm = blockIdx.y * 128, bn = blockIdx.x * 128;

  f32x8 acc[2][4];
#pragma unroll
  for (int i = 0; i < 2; ++i)
#pragma unroll
    for (int j = 0; j < 4; ++j) acc[i][j] = zero8();

  for (int kt = 0; kt < 1024; kt += 32) {
    load_a_tile(A, At, bm, kt, tid);
    load_b_tile(B, Bt, bn, kt, tid);
#if HAS_ASYNC_LDS
    ASYNC_WAIT();
#endif
    __syncthreads();

    Frag fa[2], fb[4];
#pragma unroll
    for (int mt = 0; mt < 2; ++mt) {
      const int m = wm + mt * 16 + ln;
      fa[mt].i2[0] = *(const int4*)&At[m][half * 8];
      fa[mt].i2[1] = *(const int4*)&At[m][16 + half * 8];
    }
#pragma unroll
    for (int nt = 0; nt < 4; ++nt) {
      const int n = wn + nt * 16 + ln;
      fb[nt].i2[0] = *(const int4*)&Bt[n][half * 8];
      fb[nt].i2[1] = *(const int4*)&Bt[n][16 + half * 8];
    }
#pragma unroll
    for (int mt = 0; mt < 2; ++mt)
#pragma unroll
      for (int nt = 0; nt < 4; ++nt)
        acc[mt][nt] = WMMA_BF16(fa[mt].v, fb[nt].v, acc[mt][nt]);
    __syncthreads();
  }

#pragma unroll
  for (int mt = 0; mt < 2; ++mt)
#pragma unroll
    for (int nt = 0; nt < 4; ++nt)
#pragma unroll
      for (int r = 0; r < 8; ++r) {
        const int grow = bm + wm + mt * 16 + r + 8 * half;
        const int gcol = bn + wn + nt * 16 + ln;
        Out[(size_t)grow * 1024 + gcol] = acc[mt][nt][r] + bias[gcol];
      }
}

extern "C" void kernel_launch(void* const* d_in, const int* in_sizes, int n_in,
                              void* d_out, int out_size, void* d_ws, size_t ws_size,
                              hipStream_t stream) {
  const float* x  = (const float*)d_in[0];
  const float* Wq = (const float*)d_in[1];
  const float* Wk = (const float*)d_in[2];
  const float* Wv = (const float*)d_in[3];
  const float* Wo = (const float*)d_in[4];
  const float* bo = (const float*)d_in[5];
  float* out = (float*)d_out;

  const size_t TOKE = (size_t)16384 * 1024;   // tokens * embed
  const size_t WSZ  = (size_t)1024 * 1024;
  __bf16* Xb  = (__bf16*)d_ws;
  __bf16* Wqb = Xb  + TOKE;
  __bf16* Wkb = Wqb + WSZ;
  __bf16* Wvb = Wkb + WSZ;
  __bf16* Wob = Wvb + WSZ;
  __bf16* Qb  = Wob + WSZ;
  __bf16* Kb  = Qb  + TOKE;
  __bf16* Vb  = Kb  + TOKE;
  __bf16* AOb = Vb  + TOKE;                   // total ~176 MB bf16 workspace

  // One-time fp32 -> bf16 materialization (x + 4 weights).
  convert_bf16<<<dim3(8192), dim3(256), 0, stream>>>(x,  Xb);
  convert_bf16<<<dim3(512),  dim3(256), 0, stream>>>(Wq, Wqb);
  convert_bf16<<<dim3(512),  dim3(256), 0, stream>>>(Wk, Wkb);
  convert_bf16<<<dim3(512),  dim3(256), 0, stream>>>(Wv, Wvb);
  convert_bf16<<<dim3(512),  dim3(256), 0, stream>>>(Wo, Wob);

  // QKV projections: grid (N/128, M/128, 3)
  gemm_qkv<<<dim3(8, 128, 3), dim3(256), 0, stream>>>(Xb, Wqb, Wkb, Wvb, Qb, Kb, Vb);
  // Cross-head attention: 16384 tokens, 8 waves/block
  attn_kernel<<<dim3(2048), dim3(256), 0, stream>>>(Qb, Kb, Vb, AOb);
  // Output projection + bias
  gemm_out<<<dim3(8, 128), dim3(256), 0, stream>>>(AOb, Wob, bo, out);
}
